// AttentionConv2D_21921513079523
// MI455X (gfx1250) — matrix operand order repaired and checked by reading, not compiled, not run on hardware
//
#include <hip/hip_runtime.h>
#include <stdint.h>

typedef __attribute__((ext_vector_type(16))) __bf16 v16bf;
typedef __attribute__((ext_vector_type(8)))  float  v8f;

#define CDIM 256
#define HW   4096
#define WIMG 64
#define NT   64          // pixels per workgroup (exactly one image row)
#define CP   264         // padded bf16 row stride in LDS ([pixel][c] tiles), 16B-aligned rows

struct alignas(16) U4 { unsigned x, y, z, w; };
struct alignas(16) F4 { float x, y, z, w; };
struct alignas(16) Frag { U4 a, b; };

__device__ __forceinline__ uint16_t f2bf(float f) {
  union { float f; uint32_t u; } c; c.f = f;
  uint32_t r = c.u + 0x7FFFu + ((c.u >> 16) & 1u);   // round-to-nearest-even
  return (uint16_t)(r >> 16);
}

// A-matrix 16x32 bf16 fragment: lane<16 holds K{0..7},{16..23}; lane>=16 K{8..15},{24..31}
// off is guaranteed a multiple of 8 elements (16 bytes).
__device__ __forceinline__ v16bf loadA(const uint16_t* __restrict__ w, int off) {
  const U4* p = (const U4*)__builtin_assume_aligned(w + off, 16);
  Frag fr;
  fr.a = p[0];                          // K +0..7
  fr.b = p[2];                          // K +16..23
  return __builtin_bit_cast(v16bf, fr);
}

// B-matrix 32x16 bf16 fragment: lane<16 holds K{0..7},{16..23}; lane>=16 K{8..15},{24..31}
__device__ __forceinline__ v16bf loadB(const uint16_t* __restrict__ w, int off) {
  const U4* p = (const U4*)__builtin_assume_aligned(w + off, 16);
  Frag fr;
  fr.a = p[0];                          // K +0..7
  fr.b = p[2];                          // K +16..23
  return __builtin_bit_cast(v16bf, fr);
}

// ---- prep: convert the three 256x256 weight matrices to bf16 ----
__global__ void cvt_weights(const float* __restrict__ Wq, const float* __restrict__ Wk,
                            const float* __restrict__ Wv,
                            uint16_t* __restrict__ q16, uint16_t* __restrict__ k16,
                            uint16_t* __restrict__ v16) {
  int i = blockIdx.x * blockDim.x + threadIdx.x;
  if (i < CDIM * CDIM) {
    q16[i] = f2bf(Wq[i]);
    k16[i] = f2bf(Wk[i]);
    v16[i] = f2bf(Wv[i]);
  }
}

// ---- prep: KC[p][a] = sum_c pe[c,p]*Wk[a,c] + bk[a]  (Kt for p>0 is constant; p=0 constant part) ----
__global__ void build_kc(const float* __restrict__ Wk, const float* __restrict__ bk,
                         float* __restrict__ KC) {
  int a = blockIdx.x * blockDim.x + threadIdx.x;
  if (a >= CDIM) return;
  float acc[9];
  for (int p = 0; p < 9; ++p) acc[p] = bk[a];
  const float lg = __logf(10000.0f) / 256.0f;
  for (int c = 0; c < CDIM; ++c) {
    float d = __expf(-(float)(c & ~1) * lg);
    float w = Wk[a * CDIM + c];
    for (int p = 0; p < 9; ++p) {
      float ang = (float)p * d;
      float pe = (c & 1) ? __cosf(ang) : __sinf(ang);
      acc[p] += pe * w;
    }
  }
  for (int p = 0; p < 9; ++p) KC[p * CDIM + a] = acc[p];
}

// ---- fused main kernel: Q/KX/V GEMMs (bf16 WMMA) + logits + softmax + masked scale ----
__global__ __launch_bounds__(512)
void attn_main(const float* __restrict__ x,
               const uint16_t* __restrict__ Wq16,
               const uint16_t* __restrict__ Wk16,
               const uint16_t* __restrict__ Wv16,
               const float* __restrict__ bq,
               const float* __restrict__ bv,
               const float* __restrict__ KC,
               float* __restrict__ out)
{
  extern __shared__ uint16_t tiles[];         // xT [NT][CP] then sT [NT][CP], bf16
  __shared__ float Lsum[9 * NT];
  __shared__ float att[NT];

  uint16_t* xT = tiles;
  uint16_t* sT = tiles + NT * CP;

  const int wgid = blockIdx.x;
  const int b    = wgid >> 6;                 // 8 batches
  const int h    = wgid & 63;                 // image row handled by this WG
  const int n0   = h * WIMG;
  const float* __restrict__ xb = x + (size_t)b * CDIM * HW;
  const int tid = threadIdx.x;

  for (int i = tid; i < 9 * NT; i += 512) Lsum[i] = 0.0f;

  // ---- stage tiles: thread -> (c = tid>>1, 32-pixel half), transpose f32->[pixel][c] bf16
  {
    const int c  = tid >> 1;
    const int j0 = (tid & 1) * 32;
    const float* __restrict__ rowu = xb + (size_t)c * HW + n0;
    for (int i = 0; i < 32; i += 4) {
      F4 v = *(const F4*)__builtin_assume_aligned(rowu + j0 + i, 16);
      int j = j0 + i;
      xT[(j + 0) * CP + c] = f2bf(v.x);
      xT[(j + 1) * CP + c] = f2bf(v.y);
      xT[(j + 2) * CP + c] = f2bf(v.z);
      xT[(j + 3) * CP + c] = f2bf(v.w);
    }
    // shifted tile: source pixel (h+1, j+1); zero when h==63 or j==63
    const float* __restrict__ rows = xb + (size_t)c * HW + n0 + WIMG + 1;
    for (int i = 0; i < 32; ++i) {
      int j = j0 + i;
      float v = (h < 63 && j < 63) ? rows[j] : 0.0f;
      sT[j * CP + c] = f2bf(v);
    }
  }
  __syncthreads();

  const int wave = tid >> 5;
  const int lane = tid & 31;
  const int l15  = lane & 15;
  const int hi   = lane >> 4;
  const int m0   = wave * 16;                 // this wave's 16-row A strip

  v8f accQ[4], accK[4], accV[4];
  for (int nt = 0; nt < 4; ++nt)
    for (int r = 0; r < 8; ++r) { accQ[nt][r] = 0.f; accK[nt][r] = 0.f; accV[nt][r] = 0.f; }

  const int aoffBase = (m0 + l15) * CDIM + 8 * hi;
  int boff[4];
  for (int nt = 0; nt < 4; ++nt) boff[nt] = (nt * 16 + l15) * CP + 8 * hi;

  // ---- K loop: 8 steps of 32; 12 WMMAs per step per wave
  for (int k0 = 0; k0 < CDIM; k0 += 32) {
    v16bf aq = loadA(Wq16, aoffBase + k0);
    v16bf ak = loadA(Wk16, aoffBase + k0);
    v16bf av = loadA(Wv16, aoffBase + k0);
    for (int nt = 0; nt < 4; ++nt) {
      v16bf bx = loadB(xT, boff[nt] + k0);
      v16bf bs = loadB(sT, boff[nt] + k0);
      accQ[nt] = __builtin_amdgcn_wmma_f32_16x16x32_bf16(false, aq, false, bx, (short)0, accQ[nt], false, false);
      accK[nt] = __builtin_amdgcn_wmma_f32_16x16x32_bf16(false, ak, false, bs, (short)0, accK[nt], false, false);
      accV[nt] = __builtin_amdgcn_wmma_f32_16x16x32_bf16(false, av, false, bs, (short)0, accV[nt], false, false);
    }
  }

  // ---- biases (C/D layout: VGPR r holds row m0 + r + 8*hi, column = l15)
  float bqv[8], bvv[8];
  for (int r = 0; r < 8; ++r) {
    int row = m0 + r + 8 * hi;
    bqv[r] = bq[row];
    bvv[r] = bv[row];
  }
  for (int nt = 0; nt < 4; ++nt)
    for (int r = 0; r < 8; ++r) { accQ[nt][r] += bqv[r]; accV[nt][r] += bvv[r]; }

  // ---- partial logits: l_p += Q·KC_p ; l_0 += Q⊙KX (atomic DS reduction across 16 waves)
  for (int p = 0; p < 9; ++p) {
    float kc[8];
    for (int r = 0; r < 8; ++r) kc[r] = KC[p * CDIM + m0 + r + 8 * hi];
    for (int nt = 0; nt < 4; ++nt) {
      float s = 0.f;
      for (int r = 0; r < 8; ++r) s += accQ[nt][r] * kc[r];
      atomicAdd(&Lsum[p * NT + nt * 16 + l15], s);
    }
  }
  for (int nt = 0; nt < 4; ++nt) {
    float s = 0.f;
    for (int r = 0; r < 8; ++r) s += accQ[nt][r] * accK[nt][r];
    atomicAdd(&Lsum[0 * NT + nt * 16 + l15], s);
  }
  __syncthreads();

  // ---- 9-way softmax per pixel; att0 = softmax[0] + EPS
  if (tid < NT) {
    const float scale = 0.0625f;              // 256^-0.5
    float l[9], mx = -3.4e38f;
    for (int p = 0; p < 9; ++p) { l[p] = Lsum[p * NT + tid] * scale; mx = fmaxf(mx, l[p]); }
    float se = 0.f, e0 = 0.f;
    for (int p = 0; p < 9; ++p) {
      float e = __expf(l[p] - mx);
      se += e;
      if (p == 0) e0 = e;
    }
    att[tid] = e0 / se + 1e-8f;
  }
  __syncthreads();

  // ---- out[b, c, h, w] = Vshift * att0 (Vshift masked to 0 at borders)
  float* __restrict__ ob = out + (size_t)b * CDIM * HW + n0;
  for (int nt = 0; nt < 4; ++nt) {
    int nl = nt * 16 + l15;                   // w coordinate (row-aligned tile)
    float a0 = att[nl];
    float am = (h < 63 && nl < 63) ? a0 : 0.0f;
    for (int r = 0; r < 8; ++r) {
      int row = m0 + r + 8 * hi;
      ob[(size_t)row * HW + nl] = accV[nt][r] * am;
    }
  }
}

extern "C" void kernel_launch(void* const* d_in, const int* in_sizes, int n_in,
                              void* d_out, int out_size, void* d_ws, size_t ws_size,
                              hipStream_t stream) {
  const float* x  = (const float*)d_in[0];
  const float* Wq = (const float*)d_in[1];
  const float* bq = (const float*)d_in[2];
  const float* Wk = (const float*)d_in[3];
  const float* bk = (const float*)d_in[4];
  const float* Wv = (const float*)d_in[5];
  const float* bv = (const float*)d_in[6];
  float* out = (float*)d_out;

  uint16_t* q16 = (uint16_t*)d_ws;
  uint16_t* k16 = q16 + CDIM * CDIM;
  uint16_t* v16 = k16 + CDIM * CDIM;
  float*    KC  = (float*)(v16 + CDIM * CDIM);

  cvt_weights<<<dim3(256), dim3(256), 0, stream>>>(Wq, Wk, Wv, q16, k16, v16);
  build_kc<<<dim3(1), dim3(256), 0, stream>>>(Wk, bk, KC);

  size_t lds = (size_t)2 * NT * CP * sizeof(uint16_t);   // 67584 B dynamic + small static
  attn_main<<<dim3(512), dim3(512), lds, stream>>>(x, q16, k16, v16, bq, bv, KC, out);
}
